// DKT_75033078661882
// MI455X (gfx1250) — matrix operand order repaired and checked by reading, not compiled, not run on hardware
//
#include <hip/hip_runtime.h>
#include <hip/hip_bf16.h>
#include <math.h>

// Problem constants (from reference)
#define LSTEPS 500
#define NB     128
#define CDIM   512
#define HDIM   1024
#define KCAT   1536   // CDIM + HDIM
#define G4H    4096   // 4*HDIM

typedef __attribute__((ext_vector_type(16))) __bf16 v16bf;
typedef __attribute__((ext_vector_type(8)))  float  v8f;

// Fast transcendentals: hardware v_exp_f32 / v_rcp_f32 instead of IEEE div
// sequences (v_div_scale/v_div_fmas) and tanhf libcalls.
#define LOG2E 1.44269504088896340736f

__device__ __forceinline__ float fast_sigmoid(float v) {
    // 1 / (1 + 2^(-v*log2e))
    return __builtin_amdgcn_rcpf(1.0f + __builtin_amdgcn_exp2f(v * -LOG2E));
}

__device__ __forceinline__ float fast_tanh(float v) {
    // tanh(x) = 2*sigmoid(2x) - 1
    return __builtin_fmaf(
        2.0f,
        __builtin_amdgcn_rcpf(1.0f + __builtin_amdgcn_exp2f(v * (-2.0f * LOG2E))),
        -1.0f);
}

// ---------------------------------------------------------------------------
// One-time prep kernels (re-run every kernel_launch; idempotent/deterministic)
// ---------------------------------------------------------------------------

// W_cat[m][k] = (k<512 ? W_ih[m][k] : W_hh[m][k-512]) as bf16, row-major 4096x1536
__global__ void dkt_cvt_wcat(const float* __restrict__ Wih,
                             const float* __restrict__ Whh,
                             __bf16* __restrict__ Wcat) {
    int idx = blockIdx.x * blockDim.x + threadIdx.x;
    if (idx >= G4H * KCAT) return;
    int m = idx / KCAT;
    int k = idx - m * KCAT;
    float v = (k < CDIM) ? Wih[(size_t)m * CDIM + k]
                         : Whh[(size_t)m * HDIM + (k - CDIM)];
    Wcat[idx] = (__bf16)v;
}

// W_outb[c][k] = W_out[c][k] bf16, 512x1024
__global__ void dkt_cvt_wout(const float* __restrict__ Wout,
                             __bf16* __restrict__ Woutb) {
    int idx = blockIdx.x * blockDim.x + threadIdx.x;
    if (idx >= CDIM * HDIM) return;
    Woutb[idx] = (__bf16)Wout[idx];
}

// bcat[m] = b_ih[m] + b_hh[m]
__global__ void dkt_bias(const float* __restrict__ bih,
                         const float* __restrict__ bhh,
                         float* __restrict__ bcat) {
    int idx = blockIdx.x * blockDim.x + threadIdx.x;
    if (idx < G4H) bcat[idx] = bih[idx] + bhh[idx];
}

// zero initial states (parity-0 buffers)
__global__ void dkt_init(__bf16* __restrict__ hb0,
                         float*  __restrict__ c0,
                         __bf16* __restrict__ cb0) {
    int idx = blockIdx.x * blockDim.x + threadIdx.x;
    if (idx < NB * HDIM) {
        hb0[idx] = (__bf16)0.0f;
        c0[idx]  = 0.0f;
        cb0[idx] = (__bf16)0.0f;
    }
}

// ---------------------------------------------------------------------------
// Per-timestep kernel: 768 one-wave blocks.
//   blocks [0,512)  : gate tiles -> fused LSTM cell update (writes *_nxt)
//   blocks [512,768): output head y_t = sigmoid(c_prev @ W_out^T + b_out)
// ---------------------------------------------------------------------------
__global__ __launch_bounds__(32) void dkt_step(
    const float*  __restrict__ x,       // [L][N][C] fp32
    const __bf16* __restrict__ Wcat,    // [4096][1536] bf16
    const float*  __restrict__ bcat,    // [4096]
    const __bf16* __restrict__ Woutb,   // [512][1024] bf16
    const float*  __restrict__ bout,    // [512]
    const __bf16* __restrict__ hb_cur, __bf16* __restrict__ hb_nxt,  // [N][H] bf16
    const float*  __restrict__ c_cur,  float*  __restrict__ c_nxt,   // [N][H] fp32
    const __bf16* __restrict__ cb_cur, __bf16* __restrict__ cb_nxt,  // [N][H] bf16
    float* __restrict__ out,            // [L][N][C] fp32
    int t)
{
    const int lane = threadIdx.x;       // wave32
    const int hi   = lane >> 4;         // lane half (K-group select)
    const int ln   = lane & 15;         // M (A) / N (B,C,D) index
    const int bid  = blockIdx.x;

    if (bid < 512) {
        // ---------------- gate tile: rows [n0,n0+16), H-cols [j16,j16+16) ---
        const int r  = bid >> 6;        // 0..7
        const int j  = bid & 63;        // 0..63
        const int n0 = r * 16;
        const int row = n0 + ln;
        const int j16 = j * 16;

        const float*  xrow = x + ((size_t)t * NB + row) * CDIM;
        const __bf16* hrow = hb_cur + (size_t)row * HDIM;
        // B lane ln holds W_cat row (gateBase + j16 + ln)
        const __bf16* w_i = Wcat + (size_t)(0 * HDIM + j16 + ln) * KCAT;
        const __bf16* w_f = Wcat + (size_t)(1 * HDIM + j16 + ln) * KCAT;
        const __bf16* w_g = Wcat + (size_t)(2 * HDIM + j16 + ln) * KCAT;
        const __bf16* w_o = Wcat + (size_t)(3 * HDIM + j16 + ln) * KCAT;

        v8f ci = {}, cf = {}, cg = {}, co = {};

        for (int kb = 0; kb < KCAT; kb += 32) {
            // ---- A fragment (16x32 bf16): lane=row; elements 0..15 = K kb+hi*16..
            //      (contiguous, matching B)
            v16bf a;
            if (kb < CDIM) {
                const int k0 = kb + hi * 16;
#pragma unroll
                for (int e = 0; e < 8; ++e) a[e]     = (__bf16)xrow[k0 + e];
#pragma unroll
                for (int e = 0; e < 8; ++e) a[8 + e] = (__bf16)xrow[k0 + 8 + e];
            } else {
                const int k0 = (kb - CDIM) + hi * 16;
#pragma unroll
                for (int e = 0; e < 8; ++e) a[e]     = hrow[k0 + e];
#pragma unroll
                for (int e = 0; e < 8; ++e) a[8 + e] = hrow[k0 + 8 + e];
            }

            // ---- B fragments (32x16 bf16): lane ln = column; lanes 0-15 hold
            //      K kb..kb+15, lanes 16-31 hold K kb+16..kb+31 (contiguous 32B)
            const int kw = kb + hi * 16;
            v16bf bi, bf, bg, bo;
#pragma unroll
            for (int e = 0; e < 16; ++e) bi[e] = w_i[kw + e];
#pragma unroll
            for (int e = 0; e < 16; ++e) bf[e] = w_f[kw + e];
#pragma unroll
            for (int e = 0; e < 16; ++e) bg[e] = w_g[kw + e];
#pragma unroll
            for (int e = 0; e < 16; ++e) bo[e] = w_o[kw + e];

            ci = __builtin_amdgcn_wmma_f32_16x16x32_bf16(false, a, false, bi,
                                                         (short)0, ci, false, false);
            cf = __builtin_amdgcn_wmma_f32_16x16x32_bf16(false, a, false, bf,
                                                         (short)0, cf, false, false);
            cg = __builtin_amdgcn_wmma_f32_16x16x32_bf16(false, a, false, bg,
                                                         (short)0, cg, false, false);
            co = __builtin_amdgcn_wmma_f32_16x16x32_bf16(false, a, false, bo,
                                                         (short)0, co, false, false);
        }

        // ---- fused LSTM cell update on the accumulator tile ----
        // C/D layout: element v -> row n0 + v + 8*hi, col j16 + ln
        const int colh = j16 + ln;
        const float bi_b = bcat[0 * HDIM + colh];
        const float bf_b = bcat[1 * HDIM + colh];
        const float bg_b = bcat[2 * HDIM + colh];
        const float bo_b = bcat[3 * HDIM + colh];
#pragma unroll
        for (int v = 0; v < 8; ++v) {
            const int rr = n0 + v + hi * 8;
            const size_t sidx = (size_t)rr * HDIM + colh;
            const float gi = ci[v] + bi_b;
            const float gf = cf[v] + bf_b;
            const float gg = cg[v] + bg_b;
            const float go = co[v] + bo_b;
            const float cprev = c_cur[sidx];
            const float cn = fast_sigmoid(gf) * cprev
                           + fast_sigmoid(gi) * fast_tanh(gg);
            const float hn = fast_sigmoid(go) * fast_tanh(cn);
            c_nxt[sidx]  = cn;
            cb_nxt[sidx] = (__bf16)cn;
            hb_nxt[sidx] = (__bf16)hn;
        }
    } else {
        // ---------------- output head: y_t = sigmoid(c_prev @ W_out^T + b) ---
        const int q  = bid - 512;
        const int r  = q >> 5;          // 0..7
        const int cc = q & 31;          // 0..31
        const int n0 = r * 16;
        const int row = n0 + ln;
        const int c16 = cc * 16;

        const __bf16* crow = cb_cur + (size_t)row * HDIM;
        const __bf16* wrow = Woutb + (size_t)(c16 + ln) * HDIM;

        v8f acc = {};
        for (int kb = 0; kb < HDIM; kb += 32) {
            v16bf a;
            const int k0 = kb + hi * 16;
#pragma unroll
            for (int e = 0; e < 8; ++e) a[e]     = crow[k0 + e];
#pragma unroll
            for (int e = 0; e < 8; ++e) a[8 + e] = crow[k0 + 8 + e];

            v16bf b;
            const int kw = kb + hi * 16;
#pragma unroll
            for (int e = 0; e < 16; ++e) b[e] = wrow[kw + e];

            acc = __builtin_amdgcn_wmma_f32_16x16x32_bf16(false, a, false, b,
                                                          (short)0, acc, false, false);
        }

        const int col = c16 + ln;
        const float bb = bout[col];
#pragma unroll
        for (int v = 0; v < 8; ++v) {
            const int rr = n0 + v + hi * 8;
            out[((size_t)t * NB + rr) * CDIM + col] = fast_sigmoid(acc[v] + bb);
        }
    }
}

// ---------------------------------------------------------------------------
// Host-side orchestration
// ---------------------------------------------------------------------------
extern "C" void kernel_launch(void* const* d_in, const int* in_sizes, int n_in,
                              void* d_out, int out_size, void* d_ws, size_t ws_size,
                              hipStream_t stream) {
    (void)in_sizes; (void)n_in; (void)out_size; (void)ws_size;

    const float* x    = (const float*)d_in[0];
    // d_in[1] = skills (unused by eval-mode forward)
    const float* Wih  = (const float*)d_in[2];
    const float* bih  = (const float*)d_in[3];
    const float* Whh  = (const float*)d_in[4];
    const float* bhh  = (const float*)d_in[5];
    const float* Wout = (const float*)d_in[6];
    const float* bout = (const float*)d_in[7];
    float* out = (float*)d_out;

    // Workspace carve-up (~15.7 MB total, naturally aligned regions)
    char* ws = (char*)d_ws;
    size_t off = 0;
    __bf16* Wcat  = (__bf16*)(ws + off); off += (size_t)G4H * KCAT * sizeof(__bf16);
    __bf16* Woutb = (__bf16*)(ws + off); off += (size_t)CDIM * HDIM * sizeof(__bf16);
    float*  bcat  = (float*) (ws + off); off += (size_t)G4H * sizeof(float);
    __bf16* hb[2];
    hb[0] = (__bf16*)(ws + off); off += (size_t)NB * HDIM * sizeof(__bf16);
    hb[1] = (__bf16*)(ws + off); off += (size_t)NB * HDIM * sizeof(__bf16);
    float* c32[2];
    c32[0] = (float*)(ws + off); off += (size_t)NB * HDIM * sizeof(float);
    c32[1] = (float*)(ws + off); off += (size_t)NB * HDIM * sizeof(float);
    __bf16* cb[2];
    cb[0] = (__bf16*)(ws + off); off += (size_t)NB * HDIM * sizeof(__bf16);
    cb[1] = (__bf16*)(ws + off); off += (size_t)NB * HDIM * sizeof(__bf16);

    // Prep: bf16 weight packs + fused bias + zeroed initial state
    dkt_cvt_wcat<<<(G4H * KCAT + 255) / 256, 256, 0, stream>>>(Wih, Whh, Wcat);
    dkt_cvt_wout<<<(CDIM * HDIM + 255) / 256, 256, 0, stream>>>(Wout, Woutb);
    dkt_bias<<<(G4H + 255) / 256, 256, 0, stream>>>(bih, bhh, bcat);
    dkt_init<<<(NB * HDIM + 255) / 256, 256, 0, stream>>>(hb[0], c32[0], cb[0]);

    // Recurrent chain: 500 step kernels, double-buffered state by parity
    for (int t = 0; t < LSTEPS; ++t) {
        const int cur = t & 1, nxt = cur ^ 1;
        dkt_step<<<768, 32, 0, stream>>>(
            x, Wcat, bcat, Woutb, bout,
            hb[cur], hb[nxt],
            c32[cur], c32[nxt],
            cb[cur], cb[nxt],
            out, t);
    }
}
